// AFTFullAttention_15564961481420
// MI455X (gfx1250) — compile-verified, matrix operand order checked by reading, not run on hardware
//
#include <hip/hip_runtime.h>

typedef float v2f  __attribute__((ext_vector_type(2)));
typedef float v8f  __attribute__((ext_vector_type(8)));
typedef _Float16 v16h __attribute__((ext_vector_type(16)));

#define LQ   512
#define SK   512
#define HH   8
#define DD   64
#define KDIM 64   // inner dim of u @ v^T

// ---------------------------------------------------------------------------
// Kernel 1: w[l,s] = sum_k u[l,k]*v[s,k] + mask[l,s]   (512x512, f32)
// One wave32 per 16x16 output tile; K=64 swept by V_WMMA_F32_16X16X4_F32.
// ---------------------------------------------------------------------------
__global__ __launch_bounds__(256) void aft_w_gemm(const float* __restrict__ u,
                                                  const float* __restrict__ v,
                                                  const float* __restrict__ mask,
                                                  float* __restrict__ w) {
    const int wave = threadIdx.x >> 5;
    const int lane = threadIdx.x & 31;
    const int tile = blockIdx.x * 8 + wave;          // 1024 tiles total
    const int l0   = (tile >> 5) << 4;
    const int s0   = (tile & 31) << 4;
    const int half = lane >> 4;                      // 0: lanes 0-15, 1: lanes 16-31
    const int lm   = lane & 15;

    v8f acc = {0.f, 0.f, 0.f, 0.f, 0.f, 0.f, 0.f, 0.f};

    const float* ua = u + (l0 + lm) * KDIM;          // A row (M = lm)
    const float* vb = v + (s0 + lm) * KDIM;          // B col (N = lm), B[k][n]=v[n,k]

#if __has_builtin(__builtin_amdgcn_wmma_f32_16x16x4_f32)
    // 32-bit A 16x4 layout: VGPR0 = K {0|2}, VGPR1 = K {1|3} by lane half.
    for (int kb = 0; kb < KDIM; kb += 4) {
        v2f a, b;
        a.x = ua[kb + 2 * half + 0];
        a.y = ua[kb + 2 * half + 1];
        b.x = vb[kb + 2 * half + 0];
        b.y = vb[kb + 2 * half + 1];
        acc = __builtin_amdgcn_wmma_f32_16x16x4_f32(false, a, false, b,
                                                    (short)0, acc, false, false);
    }
#else
    // Fallback: f16 16x16x32 WMMA (codegen-confirmed builtin).
    // 16-bit A 16x32 layout: half j of v16h -> K = (p%4)*2 + (j&1) + (p>=4)*16 + half*8
    for (int kb = 0; kb < KDIM; kb += 32) {
        v16h a, b;
        #pragma unroll
        for (int j = 0; j < 16; ++j) {
            int p  = j >> 1;
            int kk = ((p & 3) << 1) + (j & 1) + ((p >= 4) ? 16 : 0) + (half << 3);
            a[j] = (_Float16)ua[kb + kk];
            b[j] = (_Float16)vb[kb + kk];
        }
        acc = __builtin_amdgcn_wmma_f32_16x16x32_f16(false, a, false, b,
                                                     (short)0, acc, false, false);
    }
#endif

    // 32-bit C/D layout: VGPR r -> M = r + 8*half, N = lm.
    #pragma unroll
    for (int r = 0; r < 8; ++r) {
        int row = l0 + r + (half << 3);
        int col = s0 + lm;
        w[row * SK + col] = acc[r] + mask[row * SK + col];
    }
}

// ---------------------------------------------------------------------------
// Kernel 2: per (n,h), 16 l x 16 d tile per block. Stage w / (K+klen) / V in
// LDS in S-chunks of 256; online softmax over S per thread; gate by sigmoid(Q).
// ---------------------------------------------------------------------------
#define CHUNK 256

__global__ __launch_bounds__(256) void aft_main(const float* __restrict__ qry,
                                                const float* __restrict__ keys,
                                                const float* __restrict__ vals,
                                                const float* __restrict__ klen,
                                                const float* __restrict__ w,
                                                float* __restrict__ out) {
    __shared__ float Wsh[16 * (CHUNK + 1)];   // padded: kills 16-way bank conflict
    __shared__ float Ksh[16 * CHUNK];
    __shared__ float Vsh[16 * CHUNK];

    const int tid = threadIdx.x;
    const int l0  = blockIdx.x << 4;
    const int d0  = blockIdx.y << 4;
    const int n   = blockIdx.z / HH;
    const int h   = blockIdx.z % HH;

    const int tl = tid & 15;     // l within tile
    const int td = tid >> 4;     // d within tile

    float m  = -3.402823466e38f;
    float se = 0.f;
    float sv = 0.f;

    for (int c0 = 0; c0 < SK; c0 += CHUNK) {
        __syncthreads();   // previous chunk fully consumed
        // stage w rows (coalesced over s)
        for (int i = tid; i < 16 * CHUNK; i += 256) {
            int r = i >> 8;          // / CHUNK
            int s = i & (CHUNK - 1);
            Wsh[r * (CHUNK + 1) + s] = w[(l0 + r) * SK + (c0 + s)];
        }
        // stage K (+ length mask) and V (coalesced over d, 64B per s)
        for (int i = tid; i < 16 * CHUNK; i += 256) {
            int di = i & 15;
            int s  = i >> 4;
            int gs = c0 + s;
            int g  = ((n * SK + gs) * HH + h) * DD + d0 + di;
            Ksh[di * CHUNK + s] = keys[g] + klen[n * SK + gs];
            Vsh[di * CHUNK + s] = vals[g];
        }
        __syncthreads();

        const float* wr = Wsh + tl * (CHUNK + 1);
        const float* kr = Ksh + td * CHUNK;
        const float* vr = Vsh + td * CHUNK;

        for (int s = 0; s < CHUNK; ++s) {
            float x = kr[s] * wr[s];
            if (x > m) {               // rescale running sums (rare after warm-up)
                float c = __expf(m - x);
                se *= c;
                sv *= c;
                m = x;
            }
            float e = __expf(x - m);
            se += e;
            sv += e * vr[s];
        }
    }

    const int gi = ((n * LQ + (l0 + tl)) * HH + h) * DD + d0 + td;
    float q   = qry[gi];
    float sig = 1.f / (1.f + __expf(-q));
    out[gi] = sig * (sv / se);
}

// ---------------------------------------------------------------------------
extern "C" void kernel_launch(void* const* d_in, const int* in_sizes, int n_in,
                              void* d_out, int out_size, void* d_ws, size_t ws_size,
                              hipStream_t stream) {
    const float* queries  = (const float*)d_in[0];   // [N,L,H,D]
    const float* keys     = (const float*)d_in[1];   // [N,S,H,D]
    const float* values   = (const float*)d_in[2];   // [N,S,H,D]
    const float* attnmask = (const float*)d_in[3];   // [L,S]
    const float* key_len  = (const float*)d_in[4];   // [N,S]
    const float* u        = (const float*)d_in[5];   // [1024,64]
    const float* v        = (const float*)d_in[6];   // [1024,64]
    float*       outp     = (float*)d_out;           // [N,L,H,D]
    float*       w_ws     = (float*)d_ws;            // [512,512] scratch (1 MB)

    // Kernel 1: 1024 tiles, 8 waves/block -> 128 blocks
    aft_w_gemm<<<dim3(128), dim3(256), 0, stream>>>(u, v, attnmask, w_ws);

    // Kernel 2: grid (L/16, D/16, N*H) = (32, 4, 16)
    aft_main<<<dim3(LQ / 16, DD / 16, 2 * HH), dim3(256), 0, stream>>>(
        queries, keys, values, key_len, w_ws, outp);
}
